// CluttrVAE_36043365548130
// MI455X (gfx1250) — compile-verified, matrix operand order checked by reading, not run on hardware
//
#include <hip/hip_runtime.h>
#include <math.h>
#include <stdint.h>

typedef __bf16 bf16;
typedef __bf16 bf16x8  __attribute__((ext_vector_type(8)));
typedef __bf16 bf16x16 __attribute__((ext_vector_type(16)));
typedef float  f32x8   __attribute__((ext_vector_type(8)));
typedef unsigned int v4u __attribute__((ext_vector_type(4)));
typedef int v8i __attribute__((ext_vector_type(8)));
typedef int v4i __attribute__((ext_vector_type(4)));

#define WMMA_BF16(af, bv, acc) \
  __builtin_amdgcn_wmma_f32_16x16x32_bf16(false, (af), false, (bv), (short)0, (acc), false, false)

// ------------------------------------------------------------------
// TDM: DMA a 16-row x G-col bf16 tile (row stride = row_stride elems)
// from global memory into LDS at byte offset lds_off.
// D# per CDNA5 ISA 8.3-8.6: group0 = {count=1, lds_addr, global_addr,
// type=2}; group1 = {data_size=1(2B), tensor_dim0=G, tensor_dim1=16,
// tile_dim0=G, tile_dim1=16, tensor_dim0_stride=row_stride}.
// 6-arg builtin form (clang-23 / therock headers).
// ------------------------------------------------------------------
__device__ __forceinline__ void tdm_load_tile16(const bf16* gsrc, uint32_t lds_off,
                                                uint32_t G, uint32_t row_stride)
{
  uint64_t ga = (uint64_t)(uintptr_t)gsrc;
  v4u g0;
  g0[0] = 1u;                                              // count=1 (valid user D#)
  g0[1] = lds_off;                                         // lds_addr (bytes)
  g0[2] = (uint32_t)(ga & 0xFFFFFFFFu);                    // global_addr[31:0]
  g0[3] = (uint32_t)((ga >> 32) & 0x01FFFFFFu) | (2u << 30); // addr[56:32] | type=2
  v8i g1;
  g1[0] = (int)(1u << 16);                                 // wg_mask=0, data_size=1 (2B)
  g1[1] = (int)((G & 0xFFFFu) << 16);                      // tensor_dim0[15:0]
  g1[2] = (int)(((G >> 16) & 0xFFFFu) | (16u << 16));      // tensor_dim0 hi | tensor_dim1=16
  g1[3] = (int)((G & 0xFFFFu) << 16);                      // tile_dim0 = G
  g1[4] = (int)16;                                         // tile_dim1=16, tile_dim2=0
  g1[5] = (int)row_stride;                                 // tensor_dim0_stride lo
  g1[6] = 0;                                               // stride hi | dim1_stride lo
  g1[7] = 0;
  v4i z4 = { 0, 0, 0, 0 };
  v8i z8 = { 0, 0, 0, 0, 0, 0, 0, 0 };
  __builtin_amdgcn_tensor_load_to_lds(g0, g1, z4, z4, z8, 0);
}

// ------------------------------------------------------------------
// Zero-fill (activation slab: establishes zero padding columns)
// ------------------------------------------------------------------
__global__ void fill_zero_kernel(uint32_t* __restrict__ p, size_t n)
{
  size_t id = (size_t)blockIdx.x * blockDim.x + threadIdx.x;
  if (id < n) p[id] = 0u;
}

// ------------------------------------------------------------------
// Pack f32 weight [K,N] row-major into WMMA B-fragment tiles (bf16).
// Tile (kt,nt) at P + (nt*Ktiles+kt)*512; lane slot = half*16+col
// holds its 16 K-values contiguously (ISA 7.12.2 16-bit layout).
// K/N tails zero-padded.
// ------------------------------------------------------------------
__global__ void pack_b_kernel(const float* __restrict__ B, bf16* __restrict__ P,
                              int K, int N, int Ktiles, int Ntiles)
{
  int id = blockIdx.x * blockDim.x + threadIdx.x;
  int total = Ktiles * Ntiles * 32;
  if (id >= total) return;
  int slot = id & 31;
  int tile = id >> 5;
  int kt = tile % Ktiles;
  int nt = tile / Ktiles;
  int k0 = kt << 5, n0 = nt << 4;
  int half = slot >> 4, c = slot & 15;
  int col = n0 + c;
  bf16* dst = P + (size_t)tile * 512 + slot * 16;
#pragma unroll
  for (int j = 0; j < 16; ++j) {
    int k = k0 + ((j < 8) ? (half * 8 + j) : (16 + half * 8 + (j - 8)));
    float v = (k < K && col < N) ? B[(size_t)k * N + col] : 0.0f;
    dst[j] = (bf16)v;
  }
}

// ------------------------------------------------------------------
// Generic WMMA GEMM: C = act(A[M,Kpad](bf16) * Bpacked + bias).
// A lda multiple of 8, >= Ktiles*32, zero-padded K tail. Fast path is
// software-pipelined: next K-step's 2 A-b128 + 4 B-b128 loads are
// issued before the current 4 independent WMMAs, so loadcnt waits
// retire against loads a full WMMA batch old. Wave tile = 16x64.
// Tile validity readfirstlane'd -> scalar branches only.
// act: 0 none, 1 relu, 2 sigmoid, 3 tanh*4
// ------------------------------------------------------------------
__global__ void __launch_bounds__(128)
gemm_kernel(const bf16* __restrict__ A, int lda,
            const bf16* __restrict__ P, int Ktiles, int Ntiles,
            const float* __restrict__ bias,
            float* __restrict__ Cf, int ldc,
            bf16* __restrict__ Cb, int ldcb,
            int M, int N, int act)
{
  int lane = threadIdx.x & 31;
  int wave = __builtin_amdgcn_readfirstlane(threadIdx.x >> 5);
  int m0 = blockIdx.y << 4;
  int nt_base = (blockIdx.x * 4 + wave) * 4;
  int nvalid = Ntiles - nt_base;
  nvalid = nvalid < 0 ? 0 : (nvalid > 4 ? 4 : nvalid);
  nvalid = __builtin_amdgcn_readfirstlane(nvalid);
  if (nvalid == 0) return;   // scalar (wave-uniform) exit

  int r = lane & 15;
  int koff = (lane >> 4) << 3;
  const bf16* arow = A + (size_t)(m0 + r) * lda + koff;
  const bf16* bbase = P + (size_t)nt_base * Ktiles * 512 + (size_t)lane * 16;
  const size_t btile = (size_t)Ktiles * 512;

  f32x8 acc0 = {}, acc1 = {}, acc2 = {}, acc3 = {};

  if (nvalid == 4) {
    bf16x8 lo = *(const bf16x8*)(arow);
    bf16x8 hi = *(const bf16x8*)(arow + 16);
    const bf16* bp = bbase;
    bf16x16 b0 = *(const bf16x16*)(bp);
    bf16x16 b1 = *(const bf16x16*)(bp + btile);
    bf16x16 b2 = *(const bf16x16*)(bp + 2 * btile);
    bf16x16 b3 = *(const bf16x16*)(bp + 3 * btile);
    for (int kt = 0; kt < Ktiles - 1; ++kt) {
      bf16x16 af;
#pragma unroll
      for (int j = 0; j < 8; ++j) { af[j] = lo[j]; af[8 + j] = hi[j]; }
      arow += 32; bp += 512;
      lo = *(const bf16x8*)(arow);           // next K-step, in flight
      hi = *(const bf16x8*)(arow + 16);      // during current WMMAs
      bf16x16 n0 = *(const bf16x16*)(bp);
      bf16x16 n1 = *(const bf16x16*)(bp + btile);
      bf16x16 n2 = *(const bf16x16*)(bp + 2 * btile);
      bf16x16 n3 = *(const bf16x16*)(bp + 3 * btile);
      acc0 = WMMA_BF16(af, b0, acc0);
      acc1 = WMMA_BF16(af, b1, acc1);
      acc2 = WMMA_BF16(af, b2, acc2);
      acc3 = WMMA_BF16(af, b3, acc3);
      b0 = n0; b1 = n1; b2 = n2; b3 = n3;
    }
    bf16x16 af;
#pragma unroll
    for (int j = 0; j < 8; ++j) { af[j] = lo[j]; af[8 + j] = hi[j]; }
    acc0 = WMMA_BF16(af, b0, acc0);
    acc1 = WMMA_BF16(af, b1, acc1);
    acc2 = WMMA_BF16(af, b2, acc2);
    acc3 = WMMA_BF16(af, b3, acc3);
  } else {
    for (int kt = 0; kt < Ktiles; ++kt) {
      bf16x8 lo = *(const bf16x8*)(arow);
      bf16x8 hi = *(const bf16x8*)(arow + 16);
      arow += 32;
      bf16x16 af;
#pragma unroll
      for (int j = 0; j < 8; ++j) { af[j] = lo[j]; af[8 + j] = hi[j]; }
      const bf16* bp = bbase + (size_t)kt * 512;
      { bf16x16 b0 = *(const bf16x16*)(bp);             acc0 = WMMA_BF16(af, b0, acc0); }
      if (nvalid > 1) { bf16x16 b1 = *(const bf16x16*)(bp + btile);     acc1 = WMMA_BF16(af, b1, acc1); }
      if (nvalid > 2) { bf16x16 b2 = *(const bf16x16*)(bp + 2 * btile); acc2 = WMMA_BF16(af, b2, acc2); }
      if (nvalid > 3) { bf16x16 b3 = *(const bf16x16*)(bp + 3 * btile); acc3 = WMMA_BF16(af, b3, acc3); }
    }
  }

  int half = lane >> 4, cc = lane & 15;
  f32x8 accs[4] = { acc0, acc1, acc2, acc3 };
#pragma unroll
  for (int u = 0; u < 4; ++u) {
    if (u >= nvalid) break;                 // scalar branch
    int col = ((nt_base + u) << 4) + cc;
    if (col >= N) continue;                 // lane guard on stores only
    float bv = bias ? bias[col] : 0.0f;
#pragma unroll
    for (int rr = 0; rr < 8; ++rr) {
      int row = m0 + half * 8 + rr;
      float v = accs[u][rr] + bv;
      if (act == 1) v = fmaxf(v, 0.0f);
      else if (act == 2) v = 1.0f / (1.0f + __expf(-v));
      else if (act == 3) v = tanhf(v) * 4.0f;
      if (Cf) Cf[(size_t)row * ldc + col] = v;
      if (Cb) Cb[(size_t)row * ldcb + col] = (bf16)v;
    }
  }
}

// ------------------------------------------------------------------
// Embedding gather -> bf16 activations (padded stride)
// ------------------------------------------------------------------
__global__ void embed_kernel(const int* __restrict__ x, const float* __restrict__ emb,
                             bf16* __restrict__ out, int total, int E, int ldo)
{
  int id = blockIdx.x * blockDim.x + threadIdx.x;
  if (id >= total) return;
  int row = id / E, col = id - row * E;
  int tok = x[row];
  out[(size_t)row * ldo + col] = (bf16)emb[(size_t)tok * E + col];
}

// out = gate*f + (1-gate)*q over padded slab (pads are 0)
__global__ void hw_combine_kernel(const float* __restrict__ gate,
                                  const float* __restrict__ fgx,
                                  const float* __restrict__ q,
                                  bf16* __restrict__ out, int total)
{
  int id = blockIdx.x * blockDim.x + threadIdx.x;
  if (id >= total) return;
  float g = gate[id];
  out[id] = (bf16)(g * fgx[id] + (1.0f - g) * q[id]);
}

// z = mean + eps * exp(0.5*logvar)
__global__ void z_kernel(const float* __restrict__ mean, const float* __restrict__ logvar,
                         const float* __restrict__ eps, bf16* __restrict__ z, int total)
{
  int id = blockIdx.x * blockDim.x + threadIdx.x;
  if (id >= total) return;
  z[id] = (bf16)(mean[id] + eps[id] * __expf(0.5f * logvar[id]));
}

// ------------------------------------------------------------------
// Persistent bi-LSTM recurrence. grid.x = 2 (dir), blockDim = 512.
// LDS: gates f32[16][4H] | c f32[16][H] | h bf16[16][Hpad] | xg stage
// bf16[2][16][4H] (TDM double buffer). Per step: wave0 issues a TDM
// tile load of the NEXT step's xg into the other stage buffer (runs
// concurrently with compute); phase1 computes gates = xg_lds + h @ Wh
// with WMMA (A-fragments via aligned ds b128 loads); phase2 fused cell
// update; wave0 then s_wait_tensorcnt(0) + barrier to publish staging.
// ------------------------------------------------------------------
__global__ void __launch_bounds__(512)
lstm_kernel(const bf16* __restrict__ xg_f, const bf16* __restrict__ xg_b,
            const bf16* __restrict__ whp_f, const bf16* __restrict__ whp_b,
            bf16* __restrict__ out, int out_ld,
            bf16* __restrict__ final_out, int final_ld,
            int H, int Hpad, int T, int xg_const)
{
  extern __shared__ char smem[];
  int dir = blockIdx.x;
  const bf16* xg  = dir ? xg_b : xg_f;
  const bf16* whp = dir ? whp_b : whp_f;
  int G = 4 * H;

  float* gates = (float*)smem;                    // [16][4H]
  float* cst   = gates + 16 * G;                  // [16][H]
  bf16*  hl    = (bf16*)(cst + 16 * H);           // [16][Hpad] (tail stays 0)
  uint32_t stage_off = (uint32_t)(16u * (uint32_t)G * 4u + 16u * (uint32_t)H * 4u
                                  + 16u * (uint32_t)Hpad * 2u);
  bf16* stage = (bf16*)(smem + stage_off);        // [2][16][4H]
  uint32_t stage_bytes = (uint32_t)(16 * G) * 2u;

  for (int i = threadIdx.x; i < 16 * H; i += blockDim.x) cst[i] = 0.0f;
  for (int i = threadIdx.x; i < 16 * Hpad; i += blockDim.x) hl[i] = (bf16)0.0f;

  int lane = threadIdx.x & 31;
  int wave = __builtin_amdgcn_readfirstlane(threadIdx.x >> 5);
  int nwaves = blockDim.x >> 5;
  int ntiles = G >> 4;
  int ktiles = Hpad >> 5;
  int rr0 = lane & 15;
  int koff = (lane >> 4) << 3;
  int half = lane >> 4, cc = lane & 15;
  const bf16* hrow = hl + (size_t)rr0 * Hpad + koff;
  uint32_t row_stride = xg_const ? (uint32_t)G : (uint32_t)T * (uint32_t)G;

  // prologue: TDM-stage step 0's xg tile into buffer 0
  if (wave == 0) {
    int t0 = dir ? (T - 1) : 0;
    const bf16* gsrc = xg + (xg_const ? 0 : (size_t)t0 * G);
    tdm_load_tile16(gsrc, stage_off, (uint32_t)G, row_stride);
    __builtin_amdgcn_s_wait_tensorcnt(0);
  }
  __syncthreads();

  for (int step = 0; step < T; ++step) {
    int t = dir ? (T - 1 - step) : step;
    int buf = xg_const ? 0 : (step & 1);
    const bf16* xs = stage + (size_t)buf * (16 * G);

    // kick off next step's TDM stage (overlaps with compute below)
    if (!xg_const && wave == 0 && step + 1 < T) {
      int tn = dir ? (T - 2 - step) : (step + 1);
      tdm_load_tile16(xg + (size_t)tn * G,
                      stage_off + (uint32_t)((step + 1) & 1) * stage_bytes,
                      (uint32_t)G, row_stride);
    }

    // ---- phase 1: gates = xg_lds + h @ Wh ----
    for (int nt = wave; nt < ntiles; nt += nwaves) {   // scalar loop bound
      int col = (nt << 4) + cc;
      f32x8 acc;
#pragma unroll
      for (int rr = 0; rr < 8; ++rr)
        acc[rr] = (float)xs[(half * 8 + rr) * G + col];
      const bf16* bp = whp + ((size_t)nt * ktiles) * 512 + (size_t)lane * 16;
      for (int kt = 0; kt < ktiles; ++kt) {
        bf16x8 lo = *(const bf16x8*)(hrow + (kt << 5));        // ds_load_b128
        bf16x8 hi = *(const bf16x8*)(hrow + (kt << 5) + 16);
        bf16x16 af;
#pragma unroll
        for (int j = 0; j < 8; ++j) { af[j] = lo[j]; af[8 + j] = hi[j]; }
        bf16x16 bfv = *(const bf16x16*)(bp + (size_t)kt * 512);
        acc = WMMA_BF16(af, bfv, acc);
      }
#pragma unroll
      for (int rr = 0; rr < 8; ++rr)
        gates[(half * 8 + rr) * G + col] = acc[rr];
    }
    __syncthreads();

    // ---- phase 2: fused cell update ----
    for (int idx = threadIdx.x; idx < 16 * H; idx += blockDim.x) {
      int b = idx / H, j = idx - b * H;
      float gi = gates[b * G + j];
      float gf = gates[b * G + H + j];
      float gg = gates[b * G + 2 * H + j];
      float go = gates[b * G + 3 * H + j];
      float i_ = 1.0f / (1.0f + __expf(-gi));
      float f_ = 1.0f / (1.0f + __expf(-gf));
      float g_ = tanhf(gg);
      float o_ = 1.0f / (1.0f + __expf(-go));
      float c_ = f_ * cst[idx] + i_ * g_;
      float h_ = o_ * tanhf(c_);
      cst[idx] = c_;
      hl[b * Hpad + j] = (bf16)h_;
      if (out) out[((size_t)b * T + t) * out_ld + dir * H + j] = (bf16)h_;
      if (final_out && step == T - 1)
        final_out[(size_t)b * final_ld + dir * H + j] = (bf16)h_;
    }
    // complete + publish next step's staged tile (and h/c updates)
    if (!xg_const && wave == 0)
      __builtin_amdgcn_s_wait_tensorcnt(0);
    __syncthreads();
  }
}

// ------------------------------------------------------------------
// Host orchestration
// ------------------------------------------------------------------
static inline int ceil_div(int a, int b) { return (a + b - 1) / b; }

struct Packed { bf16* p; int Kt, Nt; };

static Packed pack_weight(const float* W, int K, int N, char*& cur, hipStream_t stream)
{
  int Kt = ceil_div(K, 32);
  int Nt = ceil_div(N, 16);
  size_t elems = (size_t)Kt * Nt * 512;
  bf16* p = (bf16*)cur;
  cur += ((elems * sizeof(bf16) + 255) & ~(size_t)255);
  int total = Kt * Nt * 32;
  pack_b_kernel<<<ceil_div(total, 256), 256, 0, stream>>>(W, p, K, N, Kt, Nt);
  return { p, Kt, Nt };
}

static void gemm(const bf16* A, int lda, const Packed& P, const float* bias,
                 float* Cf, int ldc, bf16* Cb, int ldcb,
                 int M, int N, int act, hipStream_t stream)
{
  dim3 grid(ceil_div(P.Nt, 16), M / 16);
  gemm_kernel<<<grid, 128, 0, stream>>>(A, lda, P.p, P.Kt, P.Nt, bias,
                                        Cf, ldc, Cb, ldcb, M, N, act);
}

static size_t lstm_shmem(int H, int Hpad)
{
  int G = 4 * H;
  return (size_t)16 * G * 4 + (size_t)16 * H * 4 + (size_t)16 * Hpad * 2
       + (size_t)2 * 16 * G * 2;   // TDM double-buffer stage
}

extern "C" void kernel_launch(void* const* d_in, const int* in_sizes, int n_in,
                              void* d_out, int out_size, void* d_ws, size_t ws_size,
                              hipStream_t stream)
{
  (void)in_sizes; (void)n_in; (void)out_size; (void)ws_size;
  const int B = 16, T = 256, E = 300, EH = 300, DH = 400, L = 128, V = 32000;
  const int BT = B * T;
  const int EP = 320;        // padded K for E=300 consumers (10 k-tiles)
  const int HCP = 608;       // padded K for 2*EH=600 consumers (19 k-tiles)

  // ---- inputs (setup_inputs dict order, recursive insertion order) ----
  const int*   x   = (const int*)d_in[0];
  const float* eps = (const float*)d_in[1];
  const float* emb = (const float*)d_in[2];
  int i = 3;
  struct DW { const float *W, *b; };
  DW hw1[5], hw2[5];
  for (int k = 0; k < 5; ++k) { hw1[k].W = (const float*)d_in[i++]; hw1[k].b = (const float*)d_in[i++]; }
  for (int k = 0; k < 5; ++k) { hw2[k].W = (const float*)d_in[i++]; hw2[k].b = (const float*)d_in[i++]; }
  const float *encf_Wx = (const float*)d_in[i++], *encf_Wh = (const float*)d_in[i++], *encf_b = (const float*)d_in[i++];
  const float *encb_Wx = (const float*)d_in[i++], *encb_Wh = (const float*)d_in[i++], *encb_b = (const float*)d_in[i++];
  const float *mean_W  = (const float*)d_in[i++], *mean_b  = (const float*)d_in[i++];
  const float *lv_W    = (const float*)d_in[i++], *lv_b    = (const float*)d_in[i++];
  const float *d1f_Wx  = (const float*)d_in[i++], *d1f_Wh  = (const float*)d_in[i++], *d1f_b = (const float*)d_in[i++];
  const float *d1b_Wx  = (const float*)d_in[i++], *d1b_Wh  = (const float*)d_in[i++], *d1b_b = (const float*)d_in[i++];
  const float *d2f_Wx  = (const float*)d_in[i++], *d2f_Wh  = (const float*)d_in[i++], *d2f_b = (const float*)d_in[i++];
  const float *d2b_Wx  = (const float*)d_in[i++], *d2b_Wh  = (const float*)d_in[i++], *d2b_b = (const float*)d_in[i++];
  const float *out_W   = (const float*)d_in[i++], *out_b   = (const float*)d_in[i++];

  float* logits = (float*)d_out;                         // [4096, 32000]
  float* meanO  = logits + (size_t)BT * V;               // [16, 128]
  float* lvO    = meanO + B * L;                         // [16, 128]

  // ---- workspace bump allocator ----
  char* cur = (char*)d_ws;
  auto alloc = [&](size_t bytes) -> void* {
    void* p = (void*)cur;
    cur += (bytes + 255) & ~(size_t)255;
    return p;
  };

  // ---- pack all weights into WMMA B-fragment tiles (L2-resident) ----
  Packed p_hw1[5], p_hw2[5];
  for (int k = 0; k < 5; ++k) p_hw1[k] = pack_weight(hw1[k].W, E, E, cur, stream);
  for (int k = 0; k < 5; ++k) p_hw2[k] = pack_weight(hw2[k].W, E, E, cur, stream);
  Packed p_encf_Wx = pack_weight(encf_Wx, E, 4 * EH, cur, stream);
  Packed p_encb_Wx = pack_weight(encb_Wx, E, 4 * EH, cur, stream);
  Packed p_encf_Wh = pack_weight(encf_Wh, EH, 4 * EH, cur, stream);
  Packed p_encb_Wh = pack_weight(encb_Wh, EH, 4 * EH, cur, stream);
  Packed p_mean    = pack_weight(mean_W, 2 * EH, L, cur, stream);
  Packed p_lv      = pack_weight(lv_W,   2 * EH, L, cur, stream);
  Packed p_d1f_Wx  = pack_weight(d1f_Wx, L, 4 * DH, cur, stream);
  Packed p_d1b_Wx  = pack_weight(d1b_Wx, L, 4 * DH, cur, stream);
  Packed p_d1f_Wh  = pack_weight(d1f_Wh, DH, 4 * DH, cur, stream);
  Packed p_d1b_Wh  = pack_weight(d1b_Wh, DH, 4 * DH, cur, stream);
  Packed p_d2f_Wx  = pack_weight(d2f_Wx, 2 * DH, 4 * DH, cur, stream);
  Packed p_d2b_Wx  = pack_weight(d2b_Wx, 2 * DH, 4 * DH, cur, stream);
  Packed p_d2f_Wh  = pack_weight(d2f_Wh, DH, 4 * DH, cur, stream);
  Packed p_d2b_Wh  = pack_weight(d2b_Wh, DH, 4 * DH, cur, stream);
  Packed p_out     = pack_weight(out_W, 2 * DH, V, cur, stream);

  // ---- activations (K-padded strides; slab zero-filled below) ----
  char* acts0 = cur;
  bf16* embA  = (bf16*)alloc((size_t)BT * EP * sizeof(bf16));
  bf16* gbuf  = (bf16*)alloc((size_t)BT * EP * sizeof(bf16));
  bf16* q1buf = (bf16*)alloc((size_t)BT * EP * sizeof(bf16));
  float* fgx  = (float*)alloc((size_t)BT * EP * sizeof(float));
  float* qv   = (float*)alloc((size_t)BT * EP * sizeof(float));
  float* gatev= (float*)alloc((size_t)BT * EP * sizeof(float));
  bf16* h1    = (bf16*)alloc((size_t)BT * EP * sizeof(bf16));
  bf16* h2    = (bf16*)alloc((size_t)BT * EP * sizeof(bf16));
  bf16* xg_ef = (bf16*)alloc((size_t)BT * 4 * EH * sizeof(bf16));
  bf16* xg_eb = (bf16*)alloc((size_t)BT * 4 * EH * sizeof(bf16));
  bf16* hcat  = (bf16*)alloc((size_t)B * HCP * sizeof(bf16));
  bf16* zbuf  = (bf16*)alloc((size_t)B * L * sizeof(bf16));
  bf16* xg1f  = (bf16*)alloc((size_t)B * 4 * DH * sizeof(bf16));
  bf16* xg1b  = (bf16*)alloc((size_t)B * 4 * DH * sizeof(bf16));
  bf16* d1    = (bf16*)alloc((size_t)BT * 2 * DH * sizeof(bf16));
  bf16* xg2f  = (bf16*)alloc((size_t)BT * 4 * DH * sizeof(bf16));
  bf16* xg2b  = (bf16*)alloc((size_t)BT * 4 * DH * sizeof(bf16));
  bf16* d2    = (bf16*)alloc((size_t)BT * 2 * DH * sizeof(bf16));

  {
    size_t nwords = (size_t)(cur - acts0) / 4;
    fill_zero_kernel<<<(unsigned)((nwords + 255) / 256), 256, 0, stream>>>((uint32_t*)acts0, nwords);
  }

  // ---- 1. embedding gather ----
  {
    int total = BT * E;
    embed_kernel<<<ceil_div(total, 256), 256, 0, stream>>>(x, emb, embA, total, E, EP);
  }

  // ---- 2. two highway layers (K=300 padded to 320) ----
  bf16* xin = embA;
  bf16* xouts[2] = { h1, h2 };
  for (int hwi = 0; hwi < 2; ++hwi) {
    DW* p = hwi ? hw2 : hw1;
    Packed* pp = hwi ? p_hw2 : p_hw1;
    gemm(xin, EP, pp[0], p[0].b, nullptr, 0, gbuf, EP, BT, E, 1, stream);   // g = relu
    gemm(gbuf, EP, pp[1], p[1].b, fgx, EP, nullptr, 0, BT, E, 1, stream);   // f(g) = relu
    gemm(xin, EP, pp[2], p[2].b, nullptr, 0, q1buf, EP, BT, E, 1, stream);  // relu(x@Wq1)
    gemm(q1buf, EP, pp[3], p[3].b, qv, EP, nullptr, 0, BT, E, 0, stream);   // q
    gemm(xin, EP, pp[4], p[4].b, gatev, EP, nullptr, 0, BT, E, 2, stream);  // gate = sigmoid
    int total = BT * EP;  // pads are zero in all three f32 slabs -> writes 0
    hw_combine_kernel<<<ceil_div(total, 256), 256, 0, stream>>>(gatev, fgx, qv, xouts[hwi], total);
    xin = xouts[hwi];
  }

  // ---- 3. encoder: input projections + persistent bi-LSTM ----
  gemm(h2, EP, p_encf_Wx, encf_b, nullptr, 0, xg_ef, 4 * EH, BT, 4 * EH, 0, stream);
  gemm(h2, EP, p_encb_Wx, encb_b, nullptr, 0, xg_eb, 4 * EH, BT, 4 * EH, 0, stream);
  lstm_kernel<<<2, 512, lstm_shmem(EH, 320), stream>>>(
      xg_ef, xg_eb, p_encf_Wh.p, p_encb_Wh.p,
      nullptr, 0, hcat, HCP, EH, 320, T, 0);

  // ---- 4. mean / logvar / z ----
  gemm(hcat, HCP, p_mean, mean_b, meanO, L, nullptr, 0, B, L, 3, stream);   // tanh*4
  gemm(hcat, HCP, p_lv,   lv_b,   lvO,   L, nullptr, 0, B, L, 0, stream);
  z_kernel<<<ceil_div(B * L, 256), 256, 0, stream>>>(meanO, lvO, eps, zbuf, B * L);

  // ---- 5. decoder layer 1 (constant input z over t) ----
  gemm(zbuf, L, p_d1f_Wx, d1f_b, nullptr, 0, xg1f, 4 * DH, B, 4 * DH, 0, stream);
  gemm(zbuf, L, p_d1b_Wx, d1b_b, nullptr, 0, xg1b, 4 * DH, B, 4 * DH, 0, stream);
  lstm_kernel<<<2, 512, lstm_shmem(DH, 416), stream>>>(
      xg1f, xg1b, p_d1f_Wh.p, p_d1b_Wh.p,
      d1, 2 * DH, nullptr, 0, DH, 416, T, 1);

  // ---- 6. decoder layer 2 ----
  gemm(d1, 2 * DH, p_d2f_Wx, d2f_b, nullptr, 0, xg2f, 4 * DH, BT, 4 * DH, 0, stream);
  gemm(d1, 2 * DH, p_d2b_Wx, d2b_b, nullptr, 0, xg2b, 4 * DH, BT, 4 * DH, 0, stream);
  lstm_kernel<<<2, 512, lstm_shmem(DH, 416), stream>>>(
      xg2f, xg2b, p_d2f_Wh.p, p_d2b_Wh.p,
      d2, 2 * DH, nullptr, 0, DH, 416, T, 0);

  // ---- 7. output projection: [4096,800] @ [800,32000] -> f32 logits ----
  gemm(d2, 2 * DH, p_out, out_b, logits, V, nullptr, 0, BT, V, 0, stream);
}
